// Flow_Attention_Causal_111669150275
// MI455X (gfx1250) — matrix operand order repaired and checked by reading, not compiled, not run on hardware
//
#include <hip/hip_runtime.h>
#include <hip/hip_bf16.h>

typedef __bf16 bf16_t;
typedef __attribute__((ext_vector_type(16))) __bf16 v16bf;
typedef __attribute__((ext_vector_type(8)))  __bf16 v8bf;
typedef __attribute__((ext_vector_type(4)))  __bf16 v4bf;
typedef __attribute__((ext_vector_type(8)))  float  v8f;
typedef __attribute__((ext_vector_type(4)))  float  v4f;

#define EPSF 1e-6f
#define B_SZ 4
#define NH 16
#define LSEQ 4096
#define DH 64
#define DMODEL 1024
#define MROWS (B_SZ*LSEQ)     // 16384
#define BHN (B_SZ*NH)         // 64
#define CSC 128               // scalar-scan chunk
#define NCSC (LSEQ/CSC)       // 32
#define CAT 64                // attention chunk
#define NCAT (LSEQ/CAT)       // 64
#define LDP 72                // LDS row pitch (bf16), 144B = 9*16B

static __device__ __forceinline__ v8f wmma_bf16(v16bf a, v16bf b, v8f c) {
  return __builtin_amdgcn_wmma_f32_16x16x32_bf16(false, a, false, b, (short)0, c, false, false);
}

static __device__ __forceinline__ v16bf cat8(v8bf lo, v8bf hi) {
  return __builtin_shufflevector(lo, hi, 0,1,2,3,4,5,6,7,8,9,10,11,12,13,14,15);
}

// A 16x32 (MxK) fragment from row-major LDS [m][k]: lane m=lane&15, hi=lane>>4.
// Two contiguous 8-elem runs at k0+hi*8 and k0+16+hi*8  -> 2x ds_load_b128.
static __device__ __forceinline__ v16bf ldsA(const bf16_t* base, int row0, int k0, int lane) {
  const bf16_t* p = base + (row0 + (lane & 15)) * LDP + k0 + ((lane >> 4) << 3);
  v8bf lo = *reinterpret_cast<const v8bf*>(p);
  v8bf hi = *reinterpret_cast<const v8bf*>(p + 16);
  return cat8(lo, hi);
}
// B 32x16 (KxN) fragment from TRANSPOSED LDS [n][k]: lane n=lane&15,
// two 8-elem K runs at k0+(lane>>4)*8 and k0+16+(lane>>4)*8 -> 2x ds_load_b128.
static __device__ __forceinline__ v16bf ldsBt(const bf16_t* base, int k0, int n0, int lane) {
  const bf16_t* p = base + (n0 + (lane & 15)) * LDP + k0 + ((lane >> 4) << 3);
  v8bf lo = *reinterpret_cast<const v8bf*>(p);
  v8bf hi = *reinterpret_cast<const v8bf*>(p + 16);
  return cat8(lo, hi);
}

static __device__ __forceinline__ float waveAllSum(float v) {
#pragma unroll
  for (int off = 16; off > 0; off >>= 1) v += __shfl_xor(v, off, 32);
  return v;
}

// ---- fp32 -> bf16 conversion (vectorized 4-wide) ---------------------------
__global__ void cvt_bf16_kernel(const float* __restrict__ src, bf16_t* __restrict__ dst, int n4) {
  int i = blockIdx.x * 256 + threadIdx.x;
  if (i < n4) {
    v4f s = reinterpret_cast<const v4f*>(src)[i];
    v4bf d;
#pragma unroll
    for (int e = 0; e < 4; ++e) d[e] = (bf16_t)s[e];
    reinterpret_cast<v4bf*>(dst)[i] = d;
  }
}

// ---- tiled bf16 WMMA GEMM: C = act(A[M,K] @ B[K,N] + bias) ------------------
// mode 0: headed layout [B,H,L,D] + sigmoid; mode 1: headed, no act; mode 2: flat fp32
__global__ __launch_bounds__(256) void gemm_bf16_kernel(
    const bf16_t* __restrict__ A, const bf16_t* __restrict__ Bm,
    const float* __restrict__ bias, float* __restrict__ Cout, int mode) {
  const int K = DMODEL, N = DMODEL;
  __shared__ __align__(16) bf16_t Asl[128][LDP];   // [m][k]
  __shared__ __align__(16) bf16_t Btl[128][LDP];   // [n][k] (transposed)
  const int tid = threadIdx.x;
  const int lane = tid & 31;
  const int wave = tid >> 5;
  const int m0 = blockIdx.y * 128;
  const int n0 = blockIdx.x * 128;
  const int wr = wave & 3;   // 4 row groups of 32
  const int wc = wave >> 2;  // 2 col groups of 64
  v8f acc[2][4];
#pragma unroll
  for (int i = 0; i < 2; ++i)
#pragma unroll
    for (int j = 0; j < 4; ++j) acc[i][j] = (v8f){0,0,0,0,0,0,0,0};

  for (int k0 = 0; k0 < K; k0 += 64) {
    // stage A row-major: 128x64 bf16 = 1024 x 16B chunks, b128 load + b128 store
#pragma unroll
    for (int u = tid; u < 1024; u += 256) {
      int row = u >> 3, col8 = (u & 7) << 3;
      v8bf val = *reinterpret_cast<const v8bf*>(A + (size_t)(m0 + row) * K + k0 + col8);
      *reinterpret_cast<v8bf*>(&Asl[row][col8]) = val;
    }
    // stage B transposed: global [k][n] b128 loads -> LDS [n][k] u16 scatter
#pragma unroll
    for (int u = tid; u < 1024; u += 256) {
      int krow = u >> 4, n8 = (u & 15) << 3;
      v8bf val = *reinterpret_cast<const v8bf*>(Bm + (size_t)(k0 + krow) * N + n0 + n8);
#pragma unroll
      for (int e = 0; e < 8; ++e) Btl[n8 + e][krow] = val[e];
    }
    __syncthreads();
#pragma unroll
    for (int ks = 0; ks < 64; ks += 32) {
      v16bf a0 = ldsA(&Asl[0][0], wr * 32,      ks, lane);
      v16bf a1 = ldsA(&Asl[0][0], wr * 32 + 16, ks, lane);
#pragma unroll
      for (int j = 0; j < 4; ++j) {
        v16bf bf = ldsBt(&Btl[0][0], ks, wc * 64 + j * 16, lane);
        acc[0][j] = wmma_bf16(a0, bf, acc[0][j]);
        acc[1][j] = wmma_bf16(a1, bf, acc[1][j]);
      }
    }
    __syncthreads();
  }
  const int hi = lane >> 4, ln = lane & 15;
#pragma unroll
  for (int i = 0; i < 2; ++i)
#pragma unroll
    for (int j = 0; j < 4; ++j) {
      int colg = n0 + wc * 64 + j * 16 + ln;
      float bs = bias[colg];
#pragma unroll
      for (int e = 0; e < 8; ++e) {
        int rowg = m0 + wr * 32 + i * 16 + e + hi * 8;
        float val = acc[i][j][e] + bs;
        if (mode == 0) val = 1.0f / (1.0f + __expf(-val));
        if (mode <= 1) {
          int b = rowg >> 12, l = rowg & (LSEQ - 1);
          int h = colg >> 6, d = colg & 63;
          Cout[(((size_t)(b * NH + h) * LSEQ + l) << 6) + d] = val;
        } else {
          Cout[(size_t)rowg * DMODEL + colg] = val;
        }
      }
    }
}

// ---- flow normalizers: chunked parallel scans ------------------------------
__global__ void chunk_sums_qk_kernel(const float* __restrict__ q, const float* __restrict__ k,
                                     float* __restrict__ qsum, float* __restrict__ ksum) {
  int bh = blockIdx.x >> 5, c = blockIdx.x & 31, d = threadIdx.x;
  size_t base = ((size_t)bh * LSEQ + c * CSC) * DH + d;
  float sq = 0.f, sk = 0.f;
  for (int i = 0; i < CSC; ++i) { sq += q[base + (size_t)i * DH]; sk += k[base + (size_t)i * DH]; }
  int o = (bh * NCSC + c) * DH + d;
  qsum[o] = sq; ksum[o] = sk;
}

__global__ void scan_chunks2_kernel(float* __restrict__ a, float* __restrict__ b, int nc) {
  int bh = blockIdx.x, d = threadIdx.x;
  float* pa = a + (size_t)bh * nc * DH + d;
  float* pb = b + (size_t)bh * nc * DH + d;
  float aa = 0.f, ab = 0.f;
  for (int c = 0; c < nc; ++c) {
    float t = pa[c * DH]; pa[c * DH] = aa; aa += t;
    float u = pb[c * DH]; pb[c * DH] = ab; ab += u;
  }
}

__global__ __launch_bounds__(32) void flow_pass1_kernel(
    const float* __restrict__ q, const float* __restrict__ k,
    const float* __restrict__ qexcl, const float* __restrict__ kexcl,
    float* __restrict__ sinkin, float* __restrict__ srcout, float* __restrict__ scaleq,
    float* __restrict__ qsinksum, float* __restrict__ ksrcsum) {
  int bh = blockIdx.x >> 5, c = blockIdx.x & 31, lane = threadIdx.x;
  int d0 = lane, d1 = lane + 32;
  int cb = (bh * NCSC + c) * DH;
  float cumk0 = kexcl[cb + d0], cumk1 = kexcl[cb + d1];
  float cumq0 = qexcl[cb + d0], cumq1 = qexcl[cb + d1];
  float sks0 = 0.f, sks1 = 0.f, sqs0 = 0.f, sqs1 = 0.f;
  size_t base = ((size_t)bh * LSEQ + c * CSC) * DH;
  for (int i = 0; i < CSC; ++i) {
    size_t g = base + (size_t)i * DH;
    float q0 = q[g + d0], q1 = q[g + d1], k0 = k[g + d0], k1 = k[g + d1];
    cumk0 += k0; cumk1 += k1; cumq0 += q0; cumq1 += q1;
    float dot1 = waveAllSum((q0 + EPSF) * (cumk0 + EPSF) + (q1 + EPSF) * (cumk1 + EPSF));
    float dot2 = waveAllSum((k0 + EPSF) * (cumq0 + EPSF) + (k1 + EPSF) * (cumq1 + EPSF));
    float normal = (float)(c * CSC + i + 1);
    float si = normal / dot1, so = normal / dot2;
    sks0 += k0 * so; sks1 += k1 * so; sqs0 += q0 * si; sqs1 += q1 * si;
    if (lane == 0) {
      int l = bh * LSEQ + c * CSC + i;
      sinkin[l] = si; srcout[l] = so; scaleq[l] = 1.0f / dot1;
    }
  }
  ksrcsum[cb + d0] = sks0; ksrcsum[cb + d1] = sks1;
  qsinksum[cb + d0] = sqs0; qsinksum[cb + d1] = sqs1;
}

__global__ __launch_bounds__(32) void flow_pass2_kernel(
    const float* __restrict__ q, const float* __restrict__ k,
    const float* __restrict__ qsinkexcl, const float* __restrict__ ksrcexcl,
    const float* __restrict__ sinkin, const float* __restrict__ srcout,
    float* __restrict__ allocbuf, float* __restrict__ csbuf, float* __restrict__ cssum) {
  int bh = blockIdx.x >> 5, c = blockIdx.x & 31, lane = threadIdx.x;
  int d0 = lane, d1 = lane + 32;
  int cb = (bh * NCSC + c) * DH;
  float cks0 = ksrcexcl[cb + d0], cks1 = ksrcexcl[cb + d1];
  float cqs0 = qsinkexcl[cb + d0], cqs1 = qsinkexcl[cb + d1];
  float scs = 0.f;
  size_t base = ((size_t)bh * LSEQ + c * CSC) * DH;
  for (int i = 0; i < CSC; ++i) {
    size_t g = base + (size_t)i * DH;
    int l = bh * LSEQ + c * CSC + i;
    float q0 = q[g + d0], q1 = q[g + d1], k0 = k[g + d0], k1 = k[g + d1];
    float si = sinkin[l], so = srcout[l];
    cks0 += k0 * so; cks1 += k1 * so; cqs0 += q0 * si; cqs1 += q1 * si;
    float normal = (float)(c * CSC + i + 1);
    float consink = waveAllSum((q0 + EPSF) * (cks0 + EPSF) + (q1 + EPSF) * (cks1 + EPSF)) / normal;
    float conssrc = waveAllSum((k0 + EPSF) * (cqs0 + EPSF) + (k1 + EPSF) * (cqs1 + EPSF)) / normal;
    float cs = __expf(conssrc);
    if (lane == 0) {
      allocbuf[l] = 1.0f / (1.0f + __expf(-consink));
      csbuf[l] = cs;
    }
    scs += cs;
  }
  if (lane == 0) cssum[bh * NCSC + c] = scs;
}

__global__ void scan_cssum_kernel(float* __restrict__ cssum) {
  int bh = threadIdx.x;
  float acc = 0.f;
  for (int c = 0; c < NCSC; ++c) {
    float t = cssum[bh * NCSC + c]; cssum[bh * NCSC + c] = acc; acc += t;
  }
}

__global__ __launch_bounds__(32) void comp_kernel(
    const float* __restrict__ cs, const float* __restrict__ csexcl, float* __restrict__ comp) {
  int bh = blockIdx.x >> 5, c = blockIdx.x & 31, lane = threadIdx.x;
  int lb = bh * LSEQ + c * CSC + lane * 4;
  float v0 = cs[lb], v1 = cs[lb + 1], v2 = cs[lb + 2], v3 = cs[lb + 3];
  float p0 = v0, p1 = p0 + v1, p2 = p1 + v2, p3 = p2 + v3;
  float tot = p3, run = tot;
#pragma unroll
  for (int off = 1; off < 32; off <<= 1) {
    float n = __shfl_up(run, off, 32);
    if (lane >= off) run += n;
  }
  float basev = csexcl[bh * NCSC + c] + (run - tot);
  float n0 = (float)(c * CSC + lane * 4 + 1);
  comp[lb]     = v0 / (basev + p0) * n0;
  comp[lb + 1] = v1 / (basev + p1) * (n0 + 1.f);
  comp[lb + 2] = v2 / (basev + p2) * (n0 + 2.f);
  comp[lb + 3] = v3 / (basev + p3) * (n0 + 3.f);
}

// ---- causal linear attention (chunked, WMMA) -------------------------------
// A: per-chunk KV = k^T @ v'  (grid: BH*NCAT, 256 threads)
__global__ __launch_bounds__(256) void chunk_kv_kernel(
    const float* __restrict__ k, const float* __restrict__ v,
    const float* __restrict__ comp, float* __restrict__ kvout) {
  int bh = blockIdx.x >> 6, c = blockIdx.x & 63;
  __shared__ __align__(16) bf16_t kT[CAT][LDP];   // kT[d][l] = k[l][d]
  __shared__ __align__(16) bf16_t vT[CAT][LDP];   // vT[m][l] = v'[l][m]
  int tid = threadIdx.x, lane = tid & 31, wave = tid >> 5;
  size_t base = ((size_t)bh * LSEQ + c * CAT) * DH;
#pragma unroll
  for (int i = tid; i < 1024; i += 256) {         // 4096 elems, 4-wide
    int l = i >> 4, d4 = (i & 15) << 2;
    v4f kk = *reinterpret_cast<const v4f*>(k + base + (size_t)l * DH + d4);
    v4f vv = *reinterpret_cast<const v4f*>(v + base + (size_t)l * DH + d4);
    float cp = comp[bh * LSEQ + c * CAT + l];
#pragma unroll
    for (int e = 0; e < 4; ++e) {
      kT[d4 + e][l] = (bf16_t)kk[e];
      vT[d4 + e][l] = (bf16_t)(vv[e] * cp);
    }
  }
  __syncthreads();
#pragma unroll
  for (int t = 0; t < 2; ++t) {
    int tile = wave * 2 + t, rt = tile >> 2, ct = tile & 3;
    v8f o = (v8f){0,0,0,0,0,0,0,0};
#pragma unroll
    for (int ks0 = 0; ks0 < CAT; ks0 += 32) {
      v16bf af = ldsA(&kT[0][0], rt * 16, ks0, lane);   // A[d][l]
      v16bf bf = ldsBt(&vT[0][0], ks0, ct * 16, lane);  // B[l][m] from vT[m][l]
      o = wmma_bf16(af, bf, o);
    }
    int hi = lane >> 4, ln = lane & 15;
#pragma unroll
    for (int e = 0; e < 8; ++e) {
      int d = rt * 16 + e + hi * 8, m = ct * 16 + ln;
      kvout[((size_t)(bh * NCAT + c)) * (DH * DH) + d * DH + m] = o[e];
    }
  }
}

// B: exclusive scan of KV chunks over c (per (bh,d,m) scalar scan)
__global__ void scan_kv_kernel(float* __restrict__ kv) {
  int idx = blockIdx.x * 256 + threadIdx.x;  // BH*DH*DH = 262144
  int bh = idx >> 12, dm = idx & 4095;
  float* p = kv + (size_t)bh * NCAT * (DH * DH) + dm;
  float acc = 0.f;
  for (int c = 0; c < NCAT; ++c) {
    float t = p[(size_t)c * (DH * DH)]; p[(size_t)c * (DH * DH)] = acc; acc += t;
  }
}

// C: out = (q'@KVstate + tril(q'@k^T)@v') * alloc, write bf16 [B*L, DMODEL]
__global__ __launch_bounds__(256) void attn_out_kernel(
    const float* __restrict__ q, const float* __restrict__ k, const float* __restrict__ v,
    const float* __restrict__ kv, const float* __restrict__ scaleq,
    const float* __restrict__ comp, const float* __restrict__ allocbuf,
    bf16_t* __restrict__ outbf) {
  int bh = blockIdx.x >> 6, c = blockIdx.x & 63;
  __shared__ __align__(16) bf16_t qs[CAT][LDP];    // [l][d] row-major (A operand)
  __shared__ __align__(16) bf16_t ks[CAT][LDP];    // [j][d] row-major (B of q'@k^T: [n=j][k=d])
  __shared__ __align__(16) bf16_t vT[CAT][LDP];    // vT[m][j] = v'[j][m]   (B of S@v')
  __shared__ __align__(16) bf16_t kvT[DH][LDP];    // kvT[m][d] = KV[d][m]  (B of q'@KV)
  __shared__ __align__(16) bf16_t Ss[CAT][LDP];    // [l][j] row-major (A of S@v')
  int tid = threadIdx.x, lane = tid & 31, wave = tid >> 5;
  size_t base = ((size_t)bh * LSEQ + c * CAT) * DH;
  size_t kvbase = ((size_t)(bh * NCAT + c)) * (DH * DH);
#pragma unroll
  for (int i = tid; i < 1024; i += 256) {          // 4096 elems, 4-wide
    int l = i >> 4, d4 = (i & 15) << 2;
    int lg = bh * LSEQ + c * CAT + l;
    v4f qq  = *reinterpret_cast<const v4f*>(q + base + (size_t)l * DH + d4);
    v4f kk  = *reinterpret_cast<const v4f*>(k + base + (size_t)l * DH + d4);
    v4f vv  = *reinterpret_cast<const v4f*>(v + base + (size_t)l * DH + d4);
    v4f kvv = *reinterpret_cast<const v4f*>(kv + kvbase + (size_t)l * DH + d4);
    float sq = scaleq[lg], cp = comp[lg];
    v4bf qrow, krow;
#pragma unroll
    for (int e = 0; e < 4; ++e) {
      qrow[e] = (bf16_t)(qq[e] * sq);
      krow[e] = (bf16_t)kk[e];
      vT[d4 + e][l]  = (bf16_t)(vv[e] * cp);
      kvT[d4 + e][l] = (bf16_t)kvv[e];
    }
    *reinterpret_cast<v4bf*>(&qs[l][d4]) = qrow;
    *reinterpret_cast<v4bf*>(&ks[l][d4]) = krow;
  }
  __syncthreads();
  // phase 1: S = tril(q' @ k^T)
#pragma unroll
  for (int t = 0; t < 2; ++t) {
    int tile = wave * 2 + t, rt = tile >> 2, ct = tile & 3;
    v8f s = (v8f){0,0,0,0,0,0,0,0};
#pragma unroll
    for (int ks0 = 0; ks0 < DH; ks0 += 32) {
      v16bf af = ldsA(&qs[0][0], rt * 16, ks0, lane);
      v16bf bf = ldsBt(&ks[0][0], ks0, ct * 16, lane);  // B[d][j] from ks[j][d]
      s = wmma_bf16(af, bf, s);
    }
    int hi = lane >> 4, ln = lane & 15;
#pragma unroll
    for (int e = 0; e < 8; ++e) {
      int l = rt * 16 + e + hi * 8, j = ct * 16 + ln;
      Ss[l][j] = (bf16_t)((j <= l) ? s[e] : 0.0f);
    }
  }
  __syncthreads();
  // phase 2: out = S @ v' + q' @ KVstate
#pragma unroll
  for (int t = 0; t < 2; ++t) {
    int tile = wave * 2 + t, rt = tile >> 2, ct = tile & 3;
    v8f o = (v8f){0,0,0,0,0,0,0,0};
#pragma unroll
    for (int ks0 = 0; ks0 < CAT; ks0 += 32) {
      o = wmma_bf16(ldsA(&Ss[0][0], rt * 16, ks0, lane),
                    ldsBt(&vT[0][0], ks0, ct * 16, lane), o);
      o = wmma_bf16(ldsA(&qs[0][0], rt * 16, ks0, lane),
                    ldsBt(&kvT[0][0], ks0, ct * 16, lane), o);
    }
    int hi = lane >> 4, ln = lane & 15;
    int b = bh >> 4, h = bh & 15;
#pragma unroll
    for (int e = 0; e < 8; ++e) {
      int ll = rt * 16 + e + hi * 8, m = ct * 16 + ln;
      int lg = c * CAT + ll;
      float al = allocbuf[bh * LSEQ + lg];
      outbf[((size_t)(b * LSEQ + lg)) * DMODEL + h * DH + m] = (bf16_t)(o[e] * al);
    }
  }
}

// ---- host orchestration -----------------------------------------------------
extern "C" void kernel_launch(void* const* d_in, const int* in_sizes, int n_in,
                              void* d_out, int out_size, void* d_ws, size_t ws_size,
                              hipStream_t stream) {
  const float* x  = (const float*)d_in[0];
  const float* Wq = (const float*)d_in[1];
  const float* bq = (const float*)d_in[2];
  const float* Wk = (const float*)d_in[3];
  const float* bk = (const float*)d_in[4];
  const float* Wv = (const float*)d_in[5];
  const float* bv = (const float*)d_in[6];
  const float* Wo = (const float*)d_in[7];
  const float* bo = (const float*)d_in[8];
  float* out = (float*)d_out;

  char* w = (char*)d_ws;
  size_t off = 0;
  auto take = [&](size_t bytes) -> char* {
    off = (off + 255) & ~(size_t)255;
    char* p = w + off;
    off += bytes;
    return p;
  };
  const size_t NX = (size_t)MROWS * DMODEL;        // 16.7M
  const size_t NW = (size_t)DMODEL * DMODEL;       // 1M
  const size_t NBHL = (size_t)BHN * LSEQ;          // 262144
  bf16_t* xbf  = (bf16_t*)take(NX * 2);
  bf16_t* wqbf = (bf16_t*)take(NW * 2);
  bf16_t* wkbf = (bf16_t*)take(NW * 2);
  bf16_t* wvbf = (bf16_t*)take(NW * 2);
  bf16_t* wobf = (bf16_t*)take(NW * 2);
  float* qf = (float*)take(NX * 4);
  float* kf = (float*)take(NX * 4);
  float* vf = (float*)take(NX * 4);
  float* sinkin = (float*)take(NBHL * 4);
  float* srcout = (float*)take(NBHL * 4);
  float* scaleq = (float*)take(NBHL * 4);
  float* allocb = (float*)take(NBHL * 4);
  float* csb    = (float*)take(NBHL * 4);
  float* compb  = (float*)take(NBHL * 4);
  float* qsum     = (float*)take((size_t)BHN * NCSC * DH * 4);
  float* ksum     = (float*)take((size_t)BHN * NCSC * DH * 4);
  float* qsinksum = (float*)take((size_t)BHN * NCSC * DH * 4);
  float* ksrcsum  = (float*)take((size_t)BHN * NCSC * DH * 4);
  float* cssum    = (float*)take((size_t)BHN * NCSC * 4);
  float* kvb      = (float*)take((size_t)BHN * NCAT * DH * DH * 4);
  bf16_t* attnbf  = (bf16_t*)take(NX * 2);

  // 1) bf16 conversions (4-wide)
  cvt_bf16_kernel<<<(unsigned)(NX / 1024), 256, 0, stream>>>(x, xbf, (int)(NX / 4));
  cvt_bf16_kernel<<<(unsigned)(NW / 1024), 256, 0, stream>>>(Wq, wqbf, (int)(NW / 4));
  cvt_bf16_kernel<<<(unsigned)(NW / 1024), 256, 0, stream>>>(Wk, wkbf, (int)(NW / 4));
  cvt_bf16_kernel<<<(unsigned)(NW / 1024), 256, 0, stream>>>(Wv, wvbf, (int)(NW / 4));
  cvt_bf16_kernel<<<(unsigned)(NW / 1024), 256, 0, stream>>>(Wo, wobf, (int)(NW / 4));

  // 2) projections: q,k = sigmoid(x@W+b) in [B,H,L,D]; v = x@W+b
  dim3 ggrid(DMODEL / 128, MROWS / 128);
  gemm_bf16_kernel<<<ggrid, 256, 0, stream>>>(xbf, wqbf, bq, qf, 0);
  gemm_bf16_kernel<<<ggrid, 256, 0, stream>>>(xbf, wkbf, bk, kf, 0);
  gemm_bf16_kernel<<<ggrid, 256, 0, stream>>>(xbf, wvbf, bv, vf, 1);

  // 3) flow normalizers via chunked parallel scans
  chunk_sums_qk_kernel<<<BHN * NCSC, DH, 0, stream>>>(qf, kf, qsum, ksum);
  scan_chunks2_kernel<<<BHN, DH, 0, stream>>>(qsum, ksum, NCSC);
  flow_pass1_kernel<<<BHN * NCSC, 32, 0, stream>>>(qf, kf, qsum, ksum,
                                                   sinkin, srcout, scaleq, qsinksum, ksrcsum);
  scan_chunks2_kernel<<<BHN, DH, 0, stream>>>(qsinksum, ksrcsum, NCSC);
  flow_pass2_kernel<<<BHN * NCSC, 32, 0, stream>>>(qf, kf, qsinksum, ksrcsum,
                                                   sinkin, srcout, allocb, csb, cssum);
  scan_cssum_kernel<<<1, BHN, 0, stream>>>(cssum);
  comp_kernel<<<BHN * NCSC, 32, 0, stream>>>(csb, cssum, compb);

  // 4) causal linear attention (chunk KV -> state scan -> output)
  chunk_kv_kernel<<<BHN * NCAT, 256, 0, stream>>>(kf, vf, compb, kvb);
  scan_kv_kernel<<<(BHN * DH * DH) / 256, 256, 0, stream>>>(kvb);
  attn_out_kernel<<<BHN * NCAT, 256, 0, stream>>>(qf, kf, vf, kvb, scaleq, compb, allocb, attnbf);

  // 5) output projection: out = attn @ Wo + bo (flat fp32)
  gemm_bf16_kernel<<<ggrid, 256, 0, stream>>>(attnbf, wobf, bo, out, 2);
}
